// RelationalNetwork_24550033064717
// MI455X (gfx1250) — matrix operand order repaired and checked by reading, not compiled, not run on hardware
//
#include <hip/hip_runtime.h>
#include <hip/hip_bf16.h>

typedef __bf16 bf16_t;
typedef __bf16 v16bf __attribute__((ext_vector_type(16)));
typedef __bf16 v8bf  __attribute__((ext_vector_type(8)));
typedef float  v8f   __attribute__((ext_vector_type(8)));

#define NB     32
#define OBJ    144
#define CP     26
#define HDQ    128
#define GT     256
#define TILE_M 64
#define NTILES 324   // 144*144 / 64 (exact)

// ---------------- zero the relation accumulator ----------------
__global__ void k_zero(float* __restrict__ rel) {
  rel[blockIdx.x * GT + threadIdx.x] = 0.f;
}

// ---------------- U/V precompute: per-object halves of layer 0 ----------------
// U[n,o,:] = feats[n,o] @ gW0[0:26,:]   (object in "b" slot)
// V[n,o,:] = feats[n,o] @ gW0[26:52,:]  (object in "a" slot)
__global__ void k_uv(const float* __restrict__ x, const float* __restrict__ gW0,
                     float* __restrict__ U, float* __restrict__ V) {
  int nb = blockIdx.x;                 // n*144 + o
  int n = nb / OBJ, o = nb - n * OBJ;
  int h = o / 12, w = o - h * 12;
  __shared__ float f[CP];
  int t = threadIdx.x;
  if (t < 24)       f[t]  = x[((n * 24 + t) * 12 + h) * 12 + w];
  else if (t == 24) f[24] = -1.f + (2.f / 11.f) * (float)h;   // row coord
  else if (t == 25) f[25] = -1.f + (2.f / 11.f) * (float)w;   // col coord
  __syncthreads();
  float u = 0.f, v = 0.f;
#pragma unroll
  for (int d = 0; d < CP; ++d) {
    float fd = f[d];
    u += fd * gW0[d * GT + t];
    v += fd * gW0[(CP + d) * GT + t];
  }
  U[nb * GT + t] = u;
  V[nb * GT + t] = v;
}

// ---------------- Qc[n,:] = q[n] @ gW0[52:180,:] + gb0 ----------------
__global__ void k_qc(const float* __restrict__ q, const float* __restrict__ gW0,
                     const float* __restrict__ gb0, float* __restrict__ Qc) {
  int n = blockIdx.x, t = threadIdx.x;
  __shared__ float qs[HDQ];
  if (t < HDQ) qs[t] = q[n * HDQ + t];
  __syncthreads();
  float acc = gb0[t];
  for (int d = 0; d < HDQ; ++d) acc += qs[d] * gW0[(2 * CP + d) * GT + t];
  Qc[n * GT + t] = acc;
}

// ---------------- repack gW1/2/3 (f32 KxN) into bf16 WMMA-B fragments --------
// B 32x16 bf16 layout: lane = h*16 + n ; 16 bf16 per lane: elems 0..7 = K 8h..8h+7, elems 8..15 = K 16+8h..16+8h+7.
// Fragment (kt,jt) stored lane-major so the main loop does one 32B load per lane.
__global__ void k_repack(const float* __restrict__ W1, const float* __restrict__ W2,
                         const float* __restrict__ W3, bf16_t* __restrict__ Wt) {
  int bid = blockIdx.x;            // 3 layers * 128 fragments
  int layer = bid >> 7;
  int rem = bid & 127;
  int jt = rem >> 3, kt = rem & 7;
  const float* W = (layer == 0) ? W1 : (layer == 1 ? W2 : W3);
  bf16_t* dst = Wt + layer * (GT * GT) + ((jt * 8 + kt) << 9);   // *512 elems
  for (int idx = threadIdx.x; idx < 512; idx += 256) {
    int lane = idx >> 4, i = idx & 15;
    int ncol = jt * 16 + (lane & 15);
    int K = kt * 32 + ((i >> 3) << 4) + ((lane >> 4) << 3) + (i & 7);
    dst[idx] = (bf16_t)W[K * GT + ncol];
  }
}

// helper: build a 16x32 bf16 A fragment from an LDS activation row block
__device__ __forceinline__ v16bf load_afrag(const bf16_t* arow, int half) {
  v8bf alo = *(const v8bf*)(arow + half * 8);
  v8bf ahi = *(const v8bf*)(arow + 16 + half * 8);
  return __builtin_shufflevector(alo, ahi,
      0, 1, 2, 3, 4, 5, 6, 7, 8, 9, 10, 11, 12, 13, 14, 15);
}

// ---------------- main fused g_theta + lower-triangular reduction ------------
// 8 waves/WG. Each wave owns 2 col-tiles (32 cols) and iterates all 4 row-tiles:
// 4x less B traffic per wave than a row-slab split, same 8 f32 accumulators.
__global__ void __launch_bounds__(256) k_rn_main(
    const float* __restrict__ U, const float* __restrict__ V,
    const float* __restrict__ Qc, const bf16_t* __restrict__ Wt,
    const float* __restrict__ gb1, const float* __restrict__ gb2,
    const float* __restrict__ gb3, float* __restrict__ rel) {
  const int tile = blockIdx.x, n = blockIdx.y;
  const int p0 = tile * TILE_M;

  // skip tiles entirely above the diagonal (b > a for every row)
  {
    int aA = p0 / OBJ, aB = (p0 + TILE_M - 1) / OBJ;
    bool any = false;
    { int lo = aA * OBJ, hi = lo + aA; if (p0 <= hi && p0 + TILE_M - 1 >= lo) any = true; }
    { int lo = aB * OBJ, hi = lo + aB; if (p0 <= hi && p0 + TILE_M - 1 >= lo) any = true; }
    if (!any) return;
  }

  __shared__ bf16_t hbuf[2][TILE_M * GT];    // 2 x 32KB bf16 activation ping-pong
  const int t = threadIdx.x;

  // ---- layer 0 via U + V + Qc factorization, relu, bf16 into LDS ----
  {
    float qv = Qc[n * GT + t];
    for (int r = 0; r < TILE_M; ++r) {
      int p = p0 + r, a = p / OBJ, b = p - a * OBJ;
      float f = U[(n * OBJ + b) * GT + t] + V[(n * OBJ + a) * GT + t] + qv;
      hbuf[0][r * GT + t] = (bf16_t)fmaxf(f, 0.f);
    }
  }
  __syncthreads();

  const int lane = t & 31, wv = t >> 5;
  const int jt0 = wv * 2;                    // two col-tiles owned by this wave
  const int m = lane & 15, half = lane >> 4;

  int cur = 0;
  // ---- layers 1 and 2: GEMM -> relu+bias -> bf16 LDS ----
  const float* biases01[2] = { gb1, gb2 };
  for (int L = 0; L < 2; ++L) {
    const bf16_t* Wl = Wt + L * (GT * GT);
    v8f acc[4][2] = {};
#pragma unroll
    for (int kt = 0; kt < 8; ++kt) {
      v16bf afrag[4];
#pragma unroll
      for (int rt = 0; rt < 4; ++rt)
        afrag[rt] = load_afrag(&hbuf[cur][(rt * 16 + m) * GT + kt * 32], half);
#pragma unroll
      for (int j = 0; j < 2; ++j) {
        v16bf bfrag = *(const v16bf*)(Wl + (((((jt0 + j) * 8 + kt) << 5) + lane) << 4));
#pragma unroll
        for (int rt = 0; rt < 4; ++rt)
          acc[rt][j] = __builtin_amdgcn_wmma_f32_16x16x32_bf16(
              false, afrag[rt], false, bfrag, (short)0, acc[rt][j], false, false);
      }
    }
    const float* bias = biases01[L];
#pragma unroll
    for (int j = 0; j < 2; ++j) {
      int coln = (jt0 + j) * 16 + m;
      float bv = bias[coln];
#pragma unroll
      for (int rt = 0; rt < 4; ++rt)
#pragma unroll
        for (int r = 0; r < 8; ++r) {
          int row = rt * 16 + half * 8 + r;
          hbuf[cur ^ 1][row * GT + coln] = (bf16_t)fmaxf(acc[rt][j][r] + bv, 0.f);
        }
    }
    __syncthreads();
    cur ^= 1;
  }

  // ---- layer 3: GEMM fused with masked lower-triangular reduction ----
  {
    const bf16_t* Wl = Wt + 2 * (GT * GT);
    v8f acc[4][2] = {};
#pragma unroll
    for (int kt = 0; kt < 8; ++kt) {
      v16bf afrag[4];
#pragma unroll
      for (int rt = 0; rt < 4; ++rt)
        afrag[rt] = load_afrag(&hbuf[cur][(rt * 16 + m) * GT + kt * 32], half);
#pragma unroll
      for (int j = 0; j < 2; ++j) {
        v16bf bfrag = *(const v16bf*)(Wl + (((((jt0 + j) * 8 + kt) << 5) + lane) << 4));
#pragma unroll
        for (int rt = 0; rt < 4; ++rt)
          acc[rt][j] = __builtin_amdgcn_wmma_f32_16x16x32_bf16(
              false, afrag[rt], false, bfrag, (short)0, acc[rt][j], false, false);
      }
    }
#pragma unroll
    for (int j = 0; j < 2; ++j) {
      int coln = (jt0 + j) * 16 + m;
      float bv = gb3[coln];
      float s = 0.f;
#pragma unroll
      for (int rt = 0; rt < 4; ++rt)
#pragma unroll
        for (int r = 0; r < 8; ++r) {
          int row = rt * 16 + half * 8 + r;    // this lane's 32 of the 64 rows
          int p = p0 + row, a = p / OBJ, b = p - a * OBJ;
          float v = fmaxf(acc[rt][j][r] + bv, 0.f);
          if (b <= a) s += v;
        }
      // partner lane (lane^16) holds the complementary 32 rows of the same column
      s += __shfl_xor(s, 16);
      if (half == 0) atomicAdd(&rel[n * GT + coln], s);
    }
  }
}

// ---------------- f_phi head: 256 -> 256 -> 28 ----------------
__global__ void k_fphi(const float* __restrict__ rel,
                       const float* __restrict__ fW0, const float* __restrict__ fb0,
                       const float* __restrict__ fW1, const float* __restrict__ fb1,
                       const float* __restrict__ fW2, const float* __restrict__ fb2,
                       float* __restrict__ out) {
  int n = blockIdx.x, t = threadIdx.x;
  __shared__ float s0[GT], s1[GT];
  s0[t] = rel[n * GT + t];
  __syncthreads();
  float a = fb0[t];
  for (int d = 0; d < GT; ++d) a += s0[d] * fW0[d * GT + t];
  s1[t] = fmaxf(a, 0.f);
  __syncthreads();
  a = fb1[t];
  for (int d = 0; d < GT; ++d) a += s1[d] * fW1[d * GT + t];
  __syncthreads();               // everyone done reading s0 (passed previous barrier)
  s0[t] = fmaxf(a, 0.f);
  __syncthreads();
  if (t < 28) {
    float o = fb2[t];
    for (int d = 0; d < GT; ++d) o += s0[d] * fW2[d * 28 + t];
    out[n * 28 + t] = o;
  }
}

extern "C" void kernel_launch(void* const* d_in, const int* in_sizes, int n_in,
                              void* d_out, int out_size, void* d_ws, size_t ws_size,
                              hipStream_t stream) {
  (void)in_sizes; (void)n_in; (void)out_size; (void)ws_size;
  const float* x   = (const float*)d_in[0];
  const float* q   = (const float*)d_in[1];
  const float* gW0 = (const float*)d_in[2];
  const float* gb0 = (const float*)d_in[3];
  const float* gW1 = (const float*)d_in[4];
  const float* gb1 = (const float*)d_in[5];
  const float* gW2 = (const float*)d_in[6];
  const float* gb2 = (const float*)d_in[7];
  const float* gW3 = (const float*)d_in[8];
  const float* gb3 = (const float*)d_in[9];
  const float* fW0 = (const float*)d_in[10];
  const float* fb0 = (const float*)d_in[11];
  const float* fW1 = (const float*)d_in[12];
  const float* fb1 = (const float*)d_in[13];
  const float* fW2 = (const float*)d_in[14];
  const float* fb2 = (const float*)d_in[15];
  float* out = (float*)d_out;

  // workspace layout (ws base assumed >=256B aligned; Wt offset is 32B aligned)
  float*  U   = (float*)d_ws;                 // 32*144*256 f32
  float*  V   = U  + NB * OBJ * GT;           // 32*144*256 f32
  float*  Qc  = V  + NB * OBJ * GT;           // 32*256 f32
  float*  rel = Qc + NB * GT;                 // 32*256 f32
  bf16_t* Wt  = (bf16_t*)(rel + NB * GT);     // 3*256*256 bf16 fragments

  k_zero  <<<NB,            GT,  0, stream>>>(rel);
  k_uv    <<<NB * OBJ,      GT,  0, stream>>>(x, gW0, U, V);
  k_qc    <<<NB,            GT,  0, stream>>>(q, gW0, gb0, Qc);
  k_repack<<<3 * 128,       256, 0, stream>>>(gW1, gW2, gW3, Wt);
  k_rn_main<<<dim3(NTILES, NB), 256, 0, stream>>>(U, V, Qc, Wt, gb1, gb2, gb3, rel);
  k_fphi  <<<NB,            GT,  0, stream>>>(rel, fW0, fb0, fW1, fb1, fW2, fb2, out);
}
